// CurrentHypTokenRefiner_45079976739118
// MI455X (gfx1250) — matrix operand order repaired and checked by reading, not compiled, not run on hardware
//
#include <hip/hip_runtime.h>
#include <hip/hip_bf16.h>
#include <cstddef>

#define BB   256
#define NN   2048
#define DHH  128
#define DGG  256
#define HIDD 256
#define ATT_SCALE 0.0883883476483184405f  /* 1/sqrt(128) */
#define LN_EPS 1e-5f

typedef __attribute__((ext_vector_type(16))) __bf16 v16bf;
typedef __attribute__((ext_vector_type(8)))  __bf16 v8bf;
typedef __attribute__((ext_vector_type(8)))  float  v8f;

// ---------------------------------------------------------------------------
// wave32 butterfly reductions (all lanes end with the result)
// ---------------------------------------------------------------------------
static __device__ __forceinline__ float wred_add(float v) {
#pragma unroll
  for (int m = 16; m >= 1; m >>= 1) v += __shfl_xor(v, m, 32);
  return v;
}

// ---------------------------------------------------------------------------
// fp32 -> bf16 staging (optionally transposed): dst[r*cols+c] = src[(c,r) or (r,c)]
// ---------------------------------------------------------------------------
__global__ void cvt_to_bf16(const float* __restrict__ src, __bf16* __restrict__ dst,
                            int rows, int cols, int trans) {
  int total = rows * cols;
  for (int i = blockIdx.x * blockDim.x + threadIdx.x; i < total;
       i += gridDim.x * blockDim.x) {
    int r = i / cols, c = i % cols;
    float v = trans ? src[(size_t)c * rows + r] : src[i];
    dst[i] = (__bf16)v;
  }
}

// ---------------------------------------------------------------------------
// per-row logmap0 factor: fac = artanh(clip(||x||)) / max(||x||,1e-7)
// one wave per 128-float row
// ---------------------------------------------------------------------------
__global__ void row_logmap_factor(const float* __restrict__ x, float* __restrict__ fac,
                                  int nrows) {
  int wave = threadIdx.x >> 5, lane = threadIdx.x & 31;
  int row = blockIdx.x * (blockDim.x >> 5) + wave;
  if (row >= nrows) return;
  const float4 v = *(const float4*)(x + (size_t)row * DHH + lane * 4);
  float ss = v.x * v.x + v.y * v.y + v.z * v.z + v.w * v.w;
  ss = wred_add(ss);
  float n = fmaxf(sqrtf(ss), 1e-7f);
  float a = fminf(n, 1.0f - 1e-5f);
  float t = 0.5f * __logf((1.0f + a) / (1.0f - a));   // artanh
  if (lane == 0) fac[row] = t / n;
}

// ---------------------------------------------------------------------------
// out = expmap0(A + B), rows of 128; one wave per row
// ---------------------------------------------------------------------------
__global__ void expmap_add_rows(const float* __restrict__ A, const float* __restrict__ Bv,
                                float* __restrict__ out, int nrows) {
  int wave = threadIdx.x >> 5, lane = threadIdx.x & 31;
  int row = blockIdx.x * (blockDim.x >> 5) + wave;
  if (row >= nrows) return;
  const float4 a = *(const float4*)(A  + (size_t)row * DHH + lane * 4);
  const float4 b = *(const float4*)(Bv + (size_t)row * DHH + lane * 4);
  float4 s = {a.x + b.x, a.y + b.y, a.z + b.z, a.w + b.w};
  float ss = wred_add(s.x * s.x + s.y * s.y + s.z * s.z + s.w * s.w);
  float n = fmaxf(sqrtf(ss), 1e-7f);
  float f = tanhf(n) / n;
  float4 o = {s.x * f, s.y * f, s.z * f, s.w * f};
  *(float4*)(out + (size_t)row * DHH + lane * 4) = o;
}

// ---------------------------------------------------------------------------
// u_tan = LN(logmap0(u)) * g + b  (writes f32 and bf16); one wave per row
// ---------------------------------------------------------------------------
__global__ void u_tan_ln_rows(const float* __restrict__ u,
                              const float* __restrict__ g, const float* __restrict__ be,
                              float* __restrict__ outf, __bf16* __restrict__ outb,
                              int nrows) {
  int wave = threadIdx.x >> 5, lane = threadIdx.x & 31;
  int row = blockIdx.x * (blockDim.x >> 5) + wave;
  if (row >= nrows) return;
  const float4 v = *(const float4*)(u + (size_t)row * DHH + lane * 4);
  float ss = wred_add(v.x * v.x + v.y * v.y + v.z * v.z + v.w * v.w);
  float n = fmaxf(sqrtf(ss), 1e-7f);
  float a = fminf(n, 1.0f - 1e-5f);
  float f = (0.5f * __logf((1.0f + a) / (1.0f - a))) / n;
  float4 t = {v.x * f, v.y * f, v.z * f, v.w * f};
  float mean = wred_add(t.x + t.y + t.z + t.w) * (1.0f / DHH);
  float4 d = {t.x - mean, t.y - mean, t.z - mean, t.w - mean};
  float var = wred_add(d.x * d.x + d.y * d.y + d.z * d.z + d.w * d.w) * (1.0f / DHH);
  float inv = rsqrtf(var + LN_EPS);
  int i0 = lane * 4;
  float o0 = d.x * inv * g[i0 + 0] + be[i0 + 0];
  float o1 = d.y * inv * g[i0 + 1] + be[i0 + 1];
  float o2 = d.z * inv * g[i0 + 2] + be[i0 + 2];
  float o3 = d.w * inv * g[i0 + 3] + be[i0 + 3];
  float4 o = {o0, o1, o2, o3};
  *(float4*)(outf + (size_t)row * DHH + i0) = o;
  __bf16* ob = outb + (size_t)row * DHH + i0;
  ob[0] = (__bf16)o0; ob[1] = (__bf16)o1; ob[2] = (__bf16)o2; ob[3] = (__bf16)o3;
}

// ---------------------------------------------------------------------------
// Fused attention pass (flash style, single read of demo_hyp):
//   s[n]  = (fac[n] * demo[b,n,:]) . q[b,:] * ATT_SCALE
//   out   = softmax_n(s) @ (fac[n] * demo[b,n,:])     -> bf16 (B,128)
// One block (8 waves) per batch; each wave owns rows n = wave + 8i with
// online-softmax state; combined through LDS at the end.
// ---------------------------------------------------------------------------
__global__ void attn_pool(const float* __restrict__ demo, const float* __restrict__ fac,
                          const float* __restrict__ q, __bf16* __restrict__ outb) {
  __shared__ __align__(16) float sq[DHH];
  __shared__ __align__(16) float sacc[8][DHH];
  __shared__ float sm[8], sl[8];

  int b = blockIdx.x;
  int tid = threadIdx.x, wave = tid >> 5, lane = tid & 31;
  if (tid < DHH) sq[tid] = q[(size_t)b * DHH + tid];
  __syncthreads();

  const float* base  = demo + (size_t)b * NN * DHH;
  const float* fbase = fac + (size_t)b * NN;
  const float4 qv = *(const float4*)(sq + lane * 4);

  float m = -1e30f, l = 0.0f;
  float4 acc = {0.f, 0.f, 0.f, 0.f};
  for (int n = wave; n < NN; n += 8) {
    const float* rp = base + (size_t)n * DHH + lane * 4;
    if (n + 8 < NN)
      __builtin_prefetch(base + (size_t)(n + 8) * DHH + lane * 4, 0, 1);
    const float4 v = *(const float4*)rp;
    float s = wred_add(v.x * qv.x + v.y * qv.y + v.z * qv.z + v.w * qv.w);
    float rs = fbase[n];
    float st = s * rs * ATT_SCALE;
    float mn = fmaxf(m, st);
    float c0 = __expf(m - mn);
    float p  = __expf(st - mn);
    float pw = p * rs;
    l = l * c0 + p;
    acc.x = acc.x * c0 + pw * v.x;
    acc.y = acc.y * c0 + pw * v.y;
    acc.z = acc.z * c0 + pw * v.z;
    acc.w = acc.w * c0 + pw * v.w;
    m = mn;
  }
  if (lane == 0) { sm[wave] = m; sl[wave] = l; }
  *(float4*)(&sacc[wave][lane * 4]) = acc;
  __syncthreads();

  if (tid < DHH) {
    float M = sm[0];
#pragma unroll
    for (int w = 1; w < 8; ++w) M = fmaxf(M, sm[w]);
    float denom = 0.f, sum = 0.f;
#pragma unroll
    for (int w = 0; w < 8; ++w) {
      float fw = __expf(sm[w] - M);
      denom += sl[w] * fw;
      sum   += sacc[w][tid] * fw;
    }
    outb[(size_t)b * DHH + tid] = (__bf16)(sum / denom);
  }
}

// ---------------------------------------------------------------------------
// Generic bf16 WMMA GEMM: C[M,N] = A[M,K] @ W[N,K]^T (+bias) (+Cf if acc) (gelu?)
// One wave per 16x16 tile, v_wmma_f32_16x16x32_bf16 over K.
// Fragment layouts per CDNA5 ISA 7.12.2 (16-bit A 16x32, B 32x16, f32 C 16x16).
// ---------------------------------------------------------------------------
__global__ void gemm_wmma_bf16(const __bf16* __restrict__ A, const __bf16* __restrict__ W,
                               const float* __restrict__ bias,
                               float* __restrict__ Cf, __bf16* __restrict__ Cb,
                               int M, int N, int K, int accumulate, int gelu) {
  int tm = blockIdx.x * 16, tn = blockIdx.y * 16;
  int lane = threadIdx.x & 31;
  int r = lane & 15, hi = lane >> 4;

  const __bf16* arow = A + (size_t)(tm + r) * K;
  const __bf16* wrow = W + (size_t)(tn + r) * K;

  v8f c = {0.f, 0.f, 0.f, 0.f, 0.f, 0.f, 0.f, 0.f};
  for (int k0 = 0; k0 < K; k0 += 32) {
    // A 16x32: lanes 0-15 (M=r) hold K = 0..7 & 16..23, lanes 16-31 hold K = 8..15 & 24..31
    v8bf a0 = *(const v8bf*)(arow + k0 + 8 * hi);
    v8bf a1 = *(const v8bf*)(arow + k0 + 16 + 8 * hi);
    // B 32x16: column n = tn + r; lanes 0-15 hold K = 0..7 & 16..23, lanes 16-31 hold K = 8..15 & 24..31
    v8bf b0 = *(const v8bf*)(wrow + k0 + 8 * hi);
    v8bf b1 = *(const v8bf*)(wrow + k0 + 16 + 8 * hi);
    v16bf a, bmat;
#pragma unroll
    for (int i = 0; i < 8; ++i) {
      a[i] = a0[i];  a[8 + i] = a1[i];
      bmat[i] = b0[i];  bmat[8 + i] = b1[i];
    }
    c = __builtin_amdgcn_wmma_f32_16x16x32_bf16(false, a, false, bmat,
                                                (short)0, c, false, false);
  }

  int n = tn + r;
  float bv = bias ? bias[n] : 0.0f;
#pragma unroll
  for (int i = 0; i < 8; ++i) {
    int mrow = tm + 8 * hi + i;          // C/D layout: VGPR i <-> M = 8*hi + i
    size_t idx = (size_t)mrow * N + n;
    float v = c[i] + bv;
    if (accumulate) v += Cf[idx];
    if (gelu) v = 0.5f * v * (1.0f + erff(v * 0.70710678118654752f));
    if (Cf) Cf[idx] = v;
    if (Cb) Cb[idx] = (__bf16)v;
  }
}

// ---------------------------------------------------------------------------
// Host-side orchestration
// ---------------------------------------------------------------------------
extern "C" void kernel_launch(void* const* d_in, const int* in_sizes, int n_in,
                              void* d_out, int out_size, void* d_ws, size_t ws_size,
                              hipStream_t stream) {
  const float* demo     = (const float*)d_in[0];
  const float* rho      = (const float*)d_in[1];
  const float* seed_g_w = (const float*)d_in[2];
  const float* seed_g_b = (const float*)d_in[3];
  const float* seed_d_w = (const float*)d_in[4];
  const float* pool_wq  = (const float*)d_in[5];
  const float* pool_wk  = (const float*)d_in[6];
  const float* pool_wv  = (const float*)d_in[7];
  const float* q_u_w    = (const float*)d_in[8];
  const float* q_g_w    = (const float*)d_in[9];
  const float* q_g_b    = (const float*)d_in[10];
  const float* kv_w     = (const float*)d_in[11];
  const float* mlp_w1   = (const float*)d_in[12];
  const float* mlp_b1   = (const float*)d_in[13];
  const float* mlp_w2   = (const float*)d_in[14];
  const float* mlp_b2   = (const float*)d_in[15];
  const float* ln_g     = (const float*)d_in[16];
  const float* ln_b     = (const float*)d_in[17];

  char* wsb = (char*)d_ws;
  size_t off = 0;
  auto alloc = [&](size_t bytes) -> void* {
    void* p = wsb + off;
    off = (off + bytes + 255) & ~(size_t)255;
    return p;
  };

  float*  fac       = (float*)alloc((size_t)BB * NN * 4);     // per-row logmap factors
  __bf16* rho_b     = (__bf16*)alloc((size_t)BB * DGG * 2);
  float*  q_seed_f  = (float*)alloc((size_t)BB * DHH * 4);
  __bf16* q_seed_b  = (__bf16*)alloc((size_t)BB * DHH * 2);
  __bf16* qk_b      = (__bf16*)alloc((size_t)BB * DHH * 2);
  float*  qeff_f    = (float*)alloc((size_t)BB * DHH * 4);
  __bf16* ptan_b    = (__bf16*)alloc((size_t)BB * DHH * 2);
  __bf16* pooled_b  = (__bf16*)alloc((size_t)BB * DHH * 2);
  float*  pooled2_f = (float*)alloc((size_t)BB * DHH * 4);
  float*  u_f       = (float*)alloc((size_t)BB * DHH * 4);
  float*  utan_f    = (float*)alloc((size_t)BB * DHH * 4);
  __bf16* utan_b    = (__bf16*)alloc((size_t)BB * DHH * 2);
  float*  q_f       = (float*)alloc((size_t)BB * DHH * 4);
  __bf16* q_b       = (__bf16*)alloc((size_t)BB * DHH * 2);
  __bf16* dtan_b    = (__bf16*)alloc((size_t)BB * DHH * 2);
  __bf16* delta_b   = (__bf16*)alloc((size_t)BB * DHH * 2);
  __bf16* hid_b     = (__bf16*)alloc((size_t)BB * HIDD * 2);
  float*  upd_f     = (float*)alloc((size_t)BB * DHH * 4);

  __bf16* w_seed_g   = (__bf16*)alloc((size_t)DHH * DGG * 2);
  __bf16* w_pool_wq  = (__bf16*)alloc((size_t)DHH * DHH * 2);
  __bf16* w_pool_wkT = (__bf16*)alloc((size_t)DHH * DHH * 2);
  __bf16* w_pool_wv  = (__bf16*)alloc((size_t)DHH * DHH * 2);
  __bf16* w_seed_d   = (__bf16*)alloc((size_t)DHH * DHH * 2);
  __bf16* w_q_u      = (__bf16*)alloc((size_t)DHH * DHH * 2);
  __bf16* w_q_g      = (__bf16*)alloc((size_t)DHH * DGG * 2);
  __bf16* w_kv       = (__bf16*)alloc((size_t)DHH * DHH * 2);
  __bf16* w_kvT      = (__bf16*)alloc((size_t)DHH * DHH * 2);
  __bf16* w_mlp1     = (__bf16*)alloc((size_t)HIDD * DHH * 2);
  __bf16* w_mlp2     = (__bf16*)alloc((size_t)DHH * HIDD * 2);

  auto cvt = [&](const float* s, __bf16* d, int rows, int cols, int trans) {
    int total = rows * cols;
    cvt_to_bf16<<<dim3((total + 255) / 256), dim3(256), 0, stream>>>(s, d, rows, cols, trans);
  };
  cvt(rho, rho_b, BB, DGG, 0);
  cvt(seed_g_w, w_seed_g, DHH, DGG, 0);
  cvt(pool_wq, w_pool_wq, DHH, DHH, 0);
  cvt(pool_wk, w_pool_wkT, DHH, DHH, 1);   // folded key weight (transposed)
  cvt(pool_wv, w_pool_wv, DHH, DHH, 0);
  cvt(seed_d_w, w_seed_d, DHH, DHH, 0);
  cvt(q_u_w, w_q_u, DHH, DHH, 0);
  cvt(q_g_w, w_q_g, DHH, DGG, 0);
  cvt(kv_w, w_kv, DHH, DHH, 0);
  cvt(kv_w, w_kvT, DHH, DHH, 1);           // folded query-side kv weight
  cvt(mlp_w1, w_mlp1, HIDD, DHH, 0);
  cvt(mlp_w2, w_mlp2, DHH, HIDD, 0);

  row_logmap_factor<<<dim3(BB * NN / 8), dim3(256), 0, stream>>>(demo, fac, BB * NN);

  auto gemm = [&](const __bf16* A, const __bf16* W, const float* bias,
                  float* Cf, __bf16* Cb, int M, int Ncols, int K, int acc, int act) {
    gemm_wmma_bf16<<<dim3(M / 16, Ncols / 16), dim3(32), 0, stream>>>(
        A, W, bias, Cf, Cb, M, Ncols, K, acc, act);
  };

  // seed path
  gemm(rho_b, w_seed_g, seed_g_b, q_seed_f, q_seed_b, BB, DHH, DGG, 0, 0);   // q_seed
  gemm(q_seed_b, w_pool_wq, nullptr, nullptr, qk_b, BB, DHH, DHH, 0, 0);     // qk
  gemm(qk_b, w_pool_wkT, nullptr, qeff_f, nullptr, BB, DHH, DHH, 0, 0);      // qk @ pool_wk
  attn_pool<<<dim3(BB), dim3(256), 0, stream>>>(demo, fac, qeff_f, ptan_b);  // pooled_tan
  gemm(ptan_b, w_pool_wv, nullptr, nullptr, pooled_b, BB, DHH, DHH, 0, 0);   // pooled
  gemm(pooled_b, w_seed_d, nullptr, pooled2_f, nullptr, BB, DHH, DHH, 0, 0); // @seed_d
  expmap_add_rows<<<dim3(BB / 8), dim3(256), 0, stream>>>(q_seed_f, pooled2_f, u_f, BB);

  // refinement loop (K_STEPS = 2), Kmat folded via kv_w / kv_w^T
  for (int step = 0; step < 2; ++step) {
    u_tan_ln_rows<<<dim3(BB / 8), dim3(256), 0, stream>>>(u_f, ln_g, ln_b,
                                                          utan_f, utan_b, BB);
    gemm(utan_b, w_q_u, nullptr, q_f, nullptr, BB, DHH, DHH, 0, 0);          // u_tan@q_u^T
    gemm(rho_b, w_q_g, q_g_b, q_f, q_b, BB, DHH, DGG, 1, 0);                 // += rho@q_g^T + b
    gemm(q_b, w_kvT, nullptr, qeff_f, nullptr, BB, DHH, DHH, 0, 0);          // q @ kv_w
    attn_pool<<<dim3(BB), dim3(256), 0, stream>>>(demo, fac, qeff_f, dtan_b);
    gemm(dtan_b, w_kv, nullptr, nullptr, delta_b, BB, DHH, DHH, 0, 0);       // delta
    gemm(delta_b, w_mlp1, mlp_b1, nullptr, hid_b, BB, HIDD, DHH, 0, 1);      // gelu(mlp1)
    gemm(hid_b, w_mlp2, mlp_b2, upd_f, nullptr, BB, DHH, HIDD, 0, 0);        // mlp2
    float* dst = (step == 1) ? (float*)d_out : u_f;
    expmap_add_rows<<<dim3(BB / 8), dim3(256), 0, stream>>>(utan_f, upd_f, dst, BB);
  }

  (void)in_sizes; (void)n_in; (void)out_size; (void)ws_size;
}
